// DeformableConvBlock_54219667145519
// MI455X (gfx1250) — matrix operand order repaired and checked by reading, not compiled, not run on hardware
//
#include <hip/hip_runtime.h>
#include <hip/hip_bf16.h>

typedef __attribute__((ext_vector_type(16))) _Float16 v16h;
typedef __attribute__((ext_vector_type(8)))  _Float16 v8h;
typedef __attribute__((ext_vector_type(8)))  float    v8f;

#define TM 128
#define TK 32
#define AST 40   // LDS row stride in f16 elems (80B) to stagger banks

// Problem constants (reference: B=2, C=128, H=W=128)
#define DB 2
#define DC 128
#define DH 128
#define DW 128

// ---------------- layout packing kernels ----------------

// NCHW f32 -> NHWC f16: feat = [pix][2C] (input | reference), ref16 = [pix][C]
__global__ __launch_bounds__(256) void pack_inputs_kernel(
    const float* __restrict__ xin, const float* __restrict__ xref,
    _Float16* __restrict__ feat, _Float16* __restrict__ ref16)
{
  long idx = (long)blockIdx.x * blockDim.x + threadIdx.x;  // over B*C*H*W (pow2)
  int w = (int)(idx & (DW - 1)); long t = idx >> 7;
  int h = (int)(t & (DH - 1));   t >>= 7;
  int c = (int)(t & (DC - 1));
  int b = (int)(t >> 7);
  long pix = ((long)b * DH + h) * DW + w;
  feat[pix * (2 * DC) + c]      = (_Float16)xin[idx];
  float vr = xref[idx];
  feat[pix * (2 * DC) + DC + c] = (_Float16)vr;
  ref16[pix * DC + c]           = (_Float16)vr;
}

// w[O][I][3][3] f32 -> wp[Npad][9*I] f16 with wp[co][tap*I+ci] = w[co][ci][tap]
template<int O, int I, int Npad>
__global__ __launch_bounds__(256) void pack_weight_kernel(
    const float* __restrict__ wsrc, _Float16* __restrict__ wp)
{
  constexpr int K = 9 * I;
  long idx = (long)blockIdx.x * blockDim.x + threadIdx.x;
  if (idx >= (long)Npad * K) return;
  int k  = (int)(idx % K);      // constant divisor -> mul/shift
  int co = (int)(idx / K);
  int tap = k / I;              // I is pow2
  int ci  = k % I;
  float v = (co < O) ? wsrc[((long)co * I + ci) * 9 + tap] : 0.0f;
  wp[idx] = (_Float16)v;
}

// ---------------- generic implicit-GEMM WMMA conv ----------------
// AMODE 0: A-tile = im2col(3x3, zero pad) from NHWC f16 input
// AMODE 2: A-tile = deformable bilinear samples of ref16 using f32 offsets
// OMODE 0: f16 NHWC out, ReLU          (Cstore = Cout)
// OMODE 1: f32 [pix][Cstore] out       (offset conv, Cstore=18 < TNp)
// OMODE 2: f32 NCHW out                (final result, Cstore = Cout)
// Block: 256 threads = 8 waves; tile TM=128 x TNp; wave tile (TM/NWM) x 64.
// Register-prefetch pipeline: next global tile is fetched during WMMA compute.
template<int AMODE, int OMODE, int Cin, int K, int Cstore, int TNp>
__global__ __launch_bounds__(256) void conv_wmma_kernel(
    const _Float16* __restrict__ Ain,
    const float* __restrict__ off,
    const _Float16* __restrict__ Wp,     // [TNp][K] f16 (transposed weights)
    void* __restrict__ Out)
{
  constexpr int NWN = TNp / 64;          // waves along N (1 or 2)
  constexpr int NWM = 8 / NWN;           // waves along M
  constexpr int MF  = TM / (16 * NWM);   // 16-row M-frags per wave

  __shared__ _Float16 Al[TM][AST];
  __shared__ _Float16 Bl[TNp][AST];

  const int tid  = threadIdx.x;
  const int lane = tid & 31;
  const int wave = tid >> 5;
  const int wm   = wave % NWM;
  const int wn   = wave / NWM;
  const int m0   = blockIdx.x * TM;

  // staging coordinates: 64 rows x 32 cols per pass, 8 f16 per thread
  const int lr = tid >> 2;        // 0..63
  const int lc = (tid & 3) * 8;   // 0,8,16,24

  // pixel decode for the two A staging passes (rows lr, lr+64)
  int pm[2], pb[2], ph[2], pw[2];
  #pragma unroll
  for (int i = 0; i < 2; i++) {
    const int m = m0 + lr + i * 64;
    pm[i] = m;
    pb[i] = m >> 14;               // / (128*128)
    ph[i] = (m >> 7) & (DH - 1);
    pw[i] = m & (DW - 1);
  }

  const v8f vzero = {};
  v8f acc[MF][4];
  #pragma unroll
  for (int mi = 0; mi < MF; mi++)
    #pragma unroll
    for (int t = 0; t < 4; t++) acc[mi][t] = vzero;

  const int g = lane >> 4;        // lane group (ISA 7.12.2 layouts)

  auto stageA = [&](int k0, v8h (&a_pre)[2]) {
    const int tap = k0 / Cin;     // Cin pow2: chunk never crosses a tap
    const int ci  = (k0 & (Cin - 1)) + lc;
    #pragma unroll
    for (int i = 0; i < 2; i++) {
      v8h av = {};
      if (AMODE == 0) {
        const int hh = ph[i] + tap / 3 - 1;
        const int ww = pw[i] + tap % 3 - 1;
        if (hh >= 0 && hh < DH && ww >= 0 && ww < DW)
          av = *(const v8h*)&Ain[(((long)pb[i] * DH + hh) * DW + ww) * Cin + ci];
      } else {
        const float dy = off[(long)pm[i] * 18 + 2 * tap];
        const float dx = off[(long)pm[i] * 18 + 2 * tap + 1];
        const float py = (float)(ph[i] + tap / 3 - 1) + dy;
        const float px = (float)(pw[i] + tap % 3 - 1) + dx;
        const float fy = floorf(py), fx = floorf(px);
        const int   y0 = (int)fy,    x0 = (int)fx;
        const float wy = py - fy,    wx = px - fx;
        const float wgt[4] = { (1.f - wy) * (1.f - wx), (1.f - wy) * wx,
                               wy * (1.f - wx),          wy * wx };
        float s[8] = {0.f,0.f,0.f,0.f,0.f,0.f,0.f,0.f};
        #pragma unroll
        for (int cr = 0; cr < 4; cr++) {
          const int yy = y0 + (cr >> 1);
          const int xx = x0 + (cr & 1);
          if (yy >= 0 && yy < DH && xx >= 0 && xx < DW) {
            v8h v = *(const v8h*)&Ain[(((long)pb[i] * DH + yy) * DW + xx) * Cin + ci];
            #pragma unroll
            for (int j = 0; j < 8; j++) s[j] += wgt[cr] * (float)v[j];
          }
        }
        #pragma unroll
        for (int j = 0; j < 8; j++) av[j] = (_Float16)s[j];
      }
      a_pre[i] = av;
    }
  };
  auto stageB = [&](int k0, v8h (&b_pre)[NWN]) {
    #pragma unroll
    for (int i = 0; i < NWN; i++)
      b_pre[i] = *(const v8h*)&Wp[(long)(lr + i * 64) * K + k0 + lc];
  };

  v8h a_pre[2];
  v8h b_pre[NWN];
  stageA(0, a_pre);
  stageB(0, b_pre);

  for (int k0 = 0; k0 < K; k0 += TK) {
    #pragma unroll
    for (int i = 0; i < 2; i++)   *(v8h*)&Al[lr + i * 64][lc] = a_pre[i];
    #pragma unroll
    for (int i = 0; i < NWN; i++) *(v8h*)&Bl[lr + i * 64][lc] = b_pre[i];
    __syncthreads();

    // prefetch next chunk into registers; overlaps with WMMA below
    if (k0 + TK < K) { stageA(k0 + TK, a_pre); stageB(k0 + TK, b_pre); }

    // A fragments: lanes 0-15: K[0..7]+K[16..23]; lanes 16-31: K[8..15]+K[24..31]
    v16h af[MF];
    #pragma unroll
    for (int mi = 0; mi < MF; mi++) {
      const int ml = (wm * MF + mi) * 16 + (lane & 15);
      v8h alo = *(const v8h*)&Al[ml][8 * g];
      v8h ahi = *(const v8h*)&Al[ml][16 + 8 * g];
      af[mi] = __builtin_shufflevector(alo, ahi,
          0,1,2,3,4,5,6,7,8,9,10,11,12,13,14,15);
    }
    #pragma unroll
    for (int t = 0; t < 4; t++) {
      // B fragment: lanes 0-15: K[0..7]+K[16..23]; lanes 16-31: K[8..15]+K[24..31]
      const int nl = wn * 64 + t * 16 + (lane & 15);
      v8h blo = *(const v8h*)&Bl[nl][8 * g];
      v8h bhi = *(const v8h*)&Bl[nl][16 + 8 * g];
      v16h bf = __builtin_shufflevector(blo, bhi,
          0,1,2,3,4,5,6,7,8,9,10,11,12,13,14,15);
      #pragma unroll
      for (int mi = 0; mi < MF; mi++)
        acc[mi][t] = __builtin_amdgcn_wmma_f32_16x16x32_f16(
            false, af[mi], false, bf, (short)0, acc[mi][t], false, false);
    }
    __syncthreads();
  }

  // epilogue: C/D layout: VGPR r -> M = r + 8*g, N = lane&15
  #pragma unroll
  for (int mi = 0; mi < MF; mi++) {
    #pragma unroll
    for (int t = 0; t < 4; t++) {
      const int nn = wn * 64 + t * 16 + (lane & 15);
      #pragma unroll
      for (int r = 0; r < 8; r++) {
        const int mm = m0 + (wm * MF + mi) * 16 + 8 * g + r;
        float v = acc[mi][t][r];
        if (OMODE == 0) {
          v = v > 0.f ? v : 0.f;
          ((_Float16*)Out)[(long)mm * Cstore + nn] = (_Float16)v;
        } else if (OMODE == 1) {
          if (nn < Cstore) ((float*)Out)[(long)mm * Cstore + nn] = v;
        } else {
          const int bb = mm >> 14;
          const int hh = (mm >> 7) & (DH - 1);
          const int ww = mm & (DW - 1);
          ((float*)Out)[(((long)bb * Cstore + nn) * DH + hh) * DW + ww] = v;
        }
      }
    }
  }
}

// ---------------- host launcher ----------------

extern "C" void kernel_launch(void* const* d_in, const int* in_sizes, int n_in,
                              void* d_out, int out_size, void* d_ws, size_t ws_size,
                              hipStream_t stream) {
  (void)in_sizes; (void)n_in; (void)out_size; (void)ws_size;
  const float* x_in  = (const float*)d_in[0];   // [2,128,128,128]
  const float* x_ref = (const float*)d_in[1];   // [2,128,128,128]
  const float* w1    = (const float*)d_in[2];   // [128,256,3,3]
  const float* w2    = (const float*)d_in[3];   // [128,128,3,3]
  const float* wo    = (const float*)d_in[4];   // [18,128,3,3]
  const float* wd    = (const float*)d_in[5];   // [128,128,3,3]

  const long M = (long)DB * DH * DW;            // 32768, divisible by TM

  char* ws = (char*)d_ws;
  _Float16* feat16 = (_Float16*)ws; ws += M * 256 * sizeof(_Float16);
  _Float16* ref16  = (_Float16*)ws; ws += M * 128 * sizeof(_Float16);
  _Float16* est1   = (_Float16*)ws; ws += M * 128 * sizeof(_Float16);
  _Float16* est2   = (_Float16*)ws; ws += M * 128 * sizeof(_Float16);
  float*    offs   = (float*)ws;    ws += M * 18  * sizeof(float);
  _Float16* wp1    = (_Float16*)ws; ws += 128L * (9 * 256) * sizeof(_Float16);
  _Float16* wp2    = (_Float16*)ws; ws += 128L * (9 * 128) * sizeof(_Float16);
  _Float16* wpo    = (_Float16*)ws; ws += 64L  * (9 * 128) * sizeof(_Float16);
  _Float16* wpd    = (_Float16*)ws; ws += 128L * (9 * 128) * sizeof(_Float16);

  // pack activations and weights into f16 GEMM-friendly layouts
  {
    long n = (long)DB * DC * DH * DW;
    pack_inputs_kernel<<<dim3((unsigned)(n / 256)), dim3(256), 0, stream>>>(
        x_in, x_ref, feat16, ref16);
  }
  pack_weight_kernel<128, 256, 128>
      <<<dim3((128L * 9 * 256 + 255) / 256), dim3(256), 0, stream>>>(w1, wp1);
  pack_weight_kernel<128, 128, 128>
      <<<dim3((128L * 9 * 128 + 255) / 256), dim3(256), 0, stream>>>(w2, wp2);
  pack_weight_kernel<18, 128, 64>
      <<<dim3((64L * 9 * 128 + 255) / 256), dim3(256), 0, stream>>>(wo, wpo);
  pack_weight_kernel<128, 128, 128>
      <<<dim3((128L * 9 * 128 + 255) / 256), dim3(256), 0, stream>>>(wd, wpd);

  const dim3 blk(256);
  const unsigned mblocks = (unsigned)(M / TM);  // 256

  // est1 = relu(conv3x3(feat, w1)) : Cin=256, K=2304, Cout=128
  conv_wmma_kernel<0, 0, 256, 2304, 128, 128><<<dim3(mblocks), blk, 0, stream>>>(
      feat16, nullptr, wp1, est1);
  // est2 = relu(conv3x3(est1, w2)) : Cin=128, K=1152
  conv_wmma_kernel<0, 0, 128, 1152, 128, 128><<<dim3(mblocks), blk, 0, stream>>>(
      est1, nullptr, wp2, est2);
  // offs = conv3x3(est2, wo) : Cout=18 (computed padded to 64), f32 NHWC
  conv_wmma_kernel<0, 1, 128, 1152, 18, 64><<<dim3(mblocks), blk, 0, stream>>>(
      est2, nullptr, wpo, offs);
  // out = deform_conv(ref, offs, wd) : fused bilinear sampling + GEMM, f32 NCHW
  conv_wmma_kernel<2, 2, 128, 1152, 128, 128><<<dim3(mblocks), blk, 0, stream>>>(
      ref16, offs, wpd, (float*)d_out);
}
